// DensePatchAttention_13073880449471
// MI455X (gfx1250) — matrix operand order repaired and checked by reading, not compiled, not run on hardware
//
#include <hip/hip_runtime.h>
#include <hip/hip_bf16.h>
#include <stdint.h>

// ---------------------------------------------------------------------------
// DensePatchAttention for MI455X (gfx1250, wave32, WMMA bf16 16x16x32)
//   x: [4,256,128,128] f32
//   q = 1x1 conv  -> GEMM 256x256, 65536 pixels
//   k,v = 8x8/s8 conv -> GEMM 256x16384, 1024 patches
//   attention per (b,h): N=16384, K=256 keys, c=32
// All GEMMs run on v_wmma_f32_16x16x32_bf16 (f32 accumulate).
// ---------------------------------------------------------------------------

typedef __bf16 bf16;
typedef __attribute__((ext_vector_type(16))) __bf16 v16bf;
typedef __attribute__((ext_vector_type(8)))  __bf16 v8bf;
typedef __attribute__((ext_vector_type(8)))  float  v8f;
typedef __attribute__((ext_vector_type(4)))  float  v4f;
typedef __attribute__((ext_vector_type(4)))  int    v4i;

#ifndef __has_builtin
#define __has_builtin(x) 0
#endif
#if __has_builtin(__builtin_amdgcn_global_load_async_to_lds_b128) && \
    __has_builtin(__builtin_amdgcn_s_wait_asynccnt)
#define HAVE_ASYNC_COPY 1
#else
#define HAVE_ASYNC_COPY 0
#endif

#define NBATCH 4
#define NDIM   256
#define NPIX   16384   // 128*128
#define NKEY   256     // 16*16 patches
#define NHEAD  8
#define NC     32      // channels per head

__device__ __forceinline__ v8f wmma_bf16(v16bf a, v16bf b, v8f c) {
  // D = A(16x32 bf16) * B(32x16 bf16) + C(16x16 f32)
  return __builtin_amdgcn_wmma_f32_16x16x32_bf16(false, a, false, b,
                                                 (short)0, c, false, false);
}

__device__ __forceinline__ v16bf cvt16(const float* t) {
  v16bf r;
#pragma unroll
  for (int i = 0; i < 16; ++i) r[i] = (bf16)t[i];
  return r;
}

__device__ __forceinline__ void load16(float* t, const float* p) {
  v4f a = *(const v4f*)(p);
  v4f b = *(const v4f*)(p + 4);
  v4f c = *(const v4f*)(p + 8);
  v4f d = *(const v4f*)(p + 12);
#pragma unroll
  for (int i = 0; i < 4; ++i) { t[i]=a[i]; t[4+i]=b[i]; t[8+i]=c[i]; t[12+i]=d[i]; }
}

__device__ __forceinline__ v16bf pack88(v8bf lo, v8bf hiv) {
  v16bf r;
#pragma unroll
  for (int i = 0; i < 8; ++i) { r[i]=lo[i]; r[8+i]=hiv[i]; }
  return r;
}

// ---------------------------------------------------------------------------
// Kernel 1: q = Wq(256x256) @ x + bq, stored bf16 as q[b][h][pixel][c]
// Wave tile: M=16 pixels, N=256 out-channels (16 WMMA tiles), K=256 (8 steps).
// ---------------------------------------------------------------------------
__global__ __launch_bounds__(128) void conv_q_kernel(
    const float* __restrict__ x, const float* __restrict__ Wq,
    const float* __restrict__ bq, bf16* __restrict__ qbf) {
  const int lane = threadIdx.x & 31;
  const int wv   = threadIdx.x >> 5;
  const int l15  = lane & 15;
  const int hi   = lane >> 4;      // lane half (A/B layout split)
  const int hi8  = hi * 8;

  const int gw    = blockIdx.x * 4 + wv;   // 4096 waves total
  const int bb    = gw >> 10;              // batch
  const int mbase = (gw & 1023) * 16;      // pixel tile base

  // A = x^T tile: element (m=l15 pixel, k=ci). ISA 16-bit A layout:
  //   i in [0,8): k = hi*8 + i ; i in [8,16): k = 16 + hi*8 + (i-8)
  const float* xb = x + (size_t)bb * NDIM * NPIX + (mbase + l15);
  v16bf aX[8];
#pragma unroll
  for (int ks = 0; ks < 8; ++ks) {
    float t[16];
    const int kb = ks * 32;
#pragma unroll
    for (int i = 0; i < 8; ++i) {
      t[i]     = xb[(size_t)(kb + hi8 + i)      * NPIX];
      t[8 + i] = xb[(size_t)(kb + 16 + hi8 + i) * NPIX];
    }
    aX[ks] = cvt16(t);
  }

#pragma unroll 1
  for (int nt = 0; nt < 16; ++nt) {
    const int o = nt * 16 + l15;                 // out channel (B lane = N)
    const float* wrow = Wq + (size_t)o * NDIM;   // Wq[o][ci], contiguous in ci
    v8f acc = {};
#pragma unroll
    for (int ks = 0; ks < 8; ++ks) {
      // B layout: i<8 -> k = hi*8 + i ; i>=8 -> k = 16 + hi*8 + (i-8)
      float t[16];
      {
        const float* wp = wrow + ks * 32 + hi8;
        v4f a = *(const v4f*)(wp); v4f b = *(const v4f*)(wp + 4);
        v4f c = *(const v4f*)(wp + 16); v4f d = *(const v4f*)(wp + 20);
#pragma unroll
        for (int i = 0; i < 4; ++i) { t[i]=a[i]; t[4+i]=b[i]; t[8+i]=c[i]; t[12+i]=d[i]; }
      }
      acc = wmma_bf16(aX[ks], cvt16(t), acc);
    }
    const float bias = bq[o];
    const int hh = o & 7, c = o >> 3;            // channel = c*heads + h
    bf16* qp = qbf + ((size_t)(bb * NHEAD + hh) * NPIX) * NC + c;
#pragma unroll
    for (int r = 0; r < 8; ++r) {                // D row m = r + 8*hi
      const int pix = mbase + r + hi8;
      qp[(size_t)pix * NC] = (bf16)(acc[r] + bias);
    }
  }
}

// ---------------------------------------------------------------------------
// Kernel 2: k,v = 8x8/s8 conv as GEMM with on-the-fly im2col.
// Wave tile: M=16 patch positions, N=16 out-channels, K=16384 (512 steps).
// k stored [b][h][key][c] (dots B-operand); v stored transposed [b][h][c][key]
// (PV B-operand). Both bf16.
// ---------------------------------------------------------------------------
__global__ __launch_bounds__(128) void conv_kv_kernel(
    const float* __restrict__ x,
    const float* __restrict__ Wk, const float* __restrict__ bk,
    const float* __restrict__ Wv, const float* __restrict__ bv,
    bf16* __restrict__ kbf, bf16* __restrict__ vTbf) {
  const int lane = threadIdx.x & 31;
  const int wv   = threadIdx.x >> 5;
  const int l15  = lane & 15;
  const int hi   = lane >> 4;
  const int hi8  = hi * 8;

  const int idx   = blockIdx.x * 4 + wv;  // 2048 waves: [which][b][mtile][ntile]
  const int which = idx >> 10;            // 0 = k, 1 = v
  const int rem   = idx & 1023;
  const int bb    = rem >> 8;
  const int mtile = (rem >> 4) & 15;
  const int nt    = rem & 15;
  const int mbase = mtile * 16;

  const int pos = mbase + l15;            // patch position (A lane = M row)
  const int py = pos >> 4, px = pos & 15;

  const float* W  = which ? Wv : Wk;
  const float* bp = which ? bv : bk;
  const float* xpix = x + (size_t)bb * NDIM * NPIX + px * 8;
  const int o = nt * 16 + l15;            // out channel (B lane = N)
  const float* wrow = W + (size_t)o * 16384;  // W[o][ci*64+ky*8+kx] contiguous

  v8f acc = {};
#pragma unroll 1
  for (int ks = 0; ks < 512; ++ks) {
    const int kb = ks * 32;
    __builtin_prefetch(wrow + kb + 128, 0, 1);   // global_prefetch_b8
    float ta[16];
    // A K-runs are 8-aligned -> each run has fixed (ci,ky), kx=0..7 contiguous
    {
      const int s = kb + hi8;
      const float* p = xpix + (size_t)(s >> 6) * NPIX + (py * 8 + ((s >> 3) & 7)) * 128;
      v4f a0 = *(const v4f*)p; v4f a1 = *(const v4f*)(p + 4);
#pragma unroll
      for (int i = 0; i < 4; ++i) { ta[i] = a0[i]; ta[4 + i] = a1[i]; }
    }
    {
      const int s = kb + 16 + hi8;
      const float* p = xpix + (size_t)(s >> 6) * NPIX + (py * 8 + ((s >> 3) & 7)) * 128;
      v4f a0 = *(const v4f*)p; v4f a1 = *(const v4f*)(p + 4);
#pragma unroll
      for (int i = 0; i < 4; ++i) { ta[8 + i] = a0[i]; ta[12 + i] = a1[i]; }
    }
    float tb[16];
    {   // B layout: same k order as A
      const float* wp = wrow + kb + hi8;
      v4f a = *(const v4f*)(wp); v4f b = *(const v4f*)(wp + 4);
      v4f c = *(const v4f*)(wp + 16); v4f d = *(const v4f*)(wp + 20);
#pragma unroll
      for (int i = 0; i < 4; ++i) { tb[i]=a[i]; tb[4+i]=b[i]; tb[8+i]=c[i]; tb[12+i]=d[i]; }
    }
    acc = wmma_bf16(cvt16(ta), cvt16(tb), acc);
  }

  const float bias = bp[o];
  const int hh = o & 7, c = o >> 3;
  const size_t bh = (size_t)bb * NHEAD + hh;
#pragma unroll
  for (int r = 0; r < 8; ++r) {
    const int pr = mbase + r + hi8;    // D row m = r + 8*hi
    const bf16 val = (bf16)(acc[r] + bias);
    if (which == 0) kbf [bh * (NKEY * NC) + (size_t)pr * NC + c] = val;
    else            vTbf[bh * (NKEY * NC) + (size_t)c * NKEY + pr] = val;
  }
}

// ---------------------------------------------------------------------------
// Kernel 3: fused attention per (b,h). 2 waves/block, 16 queries/wave.
// k-tile + vT-tile staged in LDS (async-to-LDS when available).
// dots: 16 WMMAs (K=32=c). softmax via shfl_xor tree. PV: 16 WMMAs.
// ---------------------------------------------------------------------------
__global__ __launch_bounds__(64) void attn_kernel(
    const bf16* __restrict__ qbf, const bf16* __restrict__ kbf,
    const bf16* __restrict__ vTbf, float* __restrict__ out) {
  __shared__ __align__(64) bf16 kT[NKEY * NC];        // [key][c]   16 KB
  __shared__ __align__(64) bf16 vT[NC * NKEY];        // [c][key]   16 KB
  __shared__ __align__(64) bf16 dots_s[2][16 * NKEY]; // per-wave raw dots 8 KB
  __shared__ float rmax_s[2][16];
  __shared__ float rsum_s[2][16];

  const int tid  = threadIdx.x;
  const int lane = tid & 31;
  const int wv   = tid >> 5;
  const int l15  = lane & 15;
  const int hi   = lane >> 4;
  const int hi8  = hi * 8;

  const int bh    = blockIdx.x >> 9;              // 32 (b,h) pairs
  const int qbase = (blockIdx.x & 511) * 32 + wv * 16;
  const int bb = bh >> 3, hh = bh & 7;

  const bf16* kg = kbf  + (size_t)bh * (NKEY * NC);
  const bf16* vg = vTbf + (size_t)bh * (NKEY * NC);

#if HAVE_ASYNC_COPY
  for (int i = tid; i < 1024; i += 64) {
    // builtin takes generic v4i* pointers (global src, lds dst, ioffset, cpol);
    // clang inserts the AS1/AS3 addrspacecasts during codegen.
    __builtin_amdgcn_global_load_async_to_lds_b128(
        (v4i*)(kg + i * 8), (v4i*)(&kT[i * 8]), 0, 0);
    __builtin_amdgcn_global_load_async_to_lds_b128(
        (v4i*)(vg + i * 8), (v4i*)(&vT[i * 8]), 0, 0);
  }
  __builtin_amdgcn_s_wait_asynccnt(0);
#else
  for (int i = tid; i < 1024; i += 64) {
    *(v8bf*)&kT[i * 8] = *(const v8bf*)(kg + i * 8);
    *(v8bf*)&vT[i * 8] = *(const v8bf*)(vg + i * 8);
  }
#endif
  __syncthreads();

  // Q as A-operand: row m = l15, k(=c) runs hi*8.. and 16+hi*8..
  const bf16* qp = qbf + ((size_t)bh * NPIX + qbase + l15) * NC;
  const v16bf aq = pack88(*(const v8bf*)(qp + hi8),
                          *(const v8bf*)(qp + 16 + hi8));

  // ---- dots = q @ k^T, track row max, spill raw dots (bf16) to LDS ----
  float rowmax[8];
#pragma unroll
  for (int r = 0; r < 8; ++r) rowmax[r] = -3.0e38f;
  bf16* myd = dots_s[wv];
#pragma unroll 1
  for (int t = 0; t < 16; ++t) {
    const bf16* kp = &kT[(t * 16 + l15) * NC + hi8]; // B: k = hi*8 + i, 16 + hi*8 + i
    v8f d = {};
    d = wmma_bf16(aq, pack88(*(const v8bf*)kp, *(const v8bf*)(kp + 16)), d);
#pragma unroll
    for (int r = 0; r < 8; ++r) {            // D: row m=r+8*hi, key=t*16+l15
      rowmax[r] = fmaxf(rowmax[r], d[r]);
      myd[(r + hi8) * NKEY + t * 16 + l15] = (bf16)d[r];
    }
  }
#pragma unroll
  for (int mask = 1; mask <= 8; mask <<= 1) {
#pragma unroll
    for (int r = 0; r < 8; ++r)
      rowmax[r] = fmaxf(rowmax[r], __shfl_xor(rowmax[r], mask, 32));
  }
  if (l15 == 0) {
#pragma unroll
    for (int r = 0; r < 8; ++r) rmax_s[wv][hi8 + r] = rowmax[r];
  }

  // ---- exp pass: build unnormalized A-operand chunks, row sums ----
  const float mrow = rmax_s[wv][l15];
  const bf16* drow = &dots_s[wv][l15 * NKEY];
  v16bf ae[8];
  float psum = 0.f;
#pragma unroll 1
  for (int kc = 0; kc < 8; ++kc) {
    v8bf e0 = *(const v8bf*)(drow + kc * 32 + hi8);
    v8bf e1 = *(const v8bf*)(drow + kc * 32 + 16 + hi8);
    v16bf a;
#pragma unroll
    for (int i = 0; i < 8; ++i) {
      const float f0 = __expf((float)e0[i] - mrow);
      const float f1 = __expf((float)e1[i] - mrow);
      psum += f0 + f1;
      a[i] = (bf16)f0; a[8 + i] = (bf16)f1;
    }
    ae[kc] = a;
  }
  psum += __shfl_xor(psum, 16, 32);  // two lane-halves cover a full row
  if (hi == 0) rsum_s[wv][l15] = psum;

  // ---- out = (e @ v) / rowsum, rearrange B h (HW) c -> B (c h) H W ----
  float rsinv[8];
#pragma unroll
  for (int r = 0; r < 8; ++r) rsinv[r] = 1.0f / rsum_s[wv][r + hi8];
#pragma unroll 1
  for (int nt = 0; nt < 2; ++nt) {
    const int c = nt * 16 + l15;               // B lane = N = c
    const bf16* vp = &vT[c * NKEY];
    v8f acc = {};
#pragma unroll
    for (int kc = 0; kc < 8; ++kc) {           // B: kk = kc*32 + hi*8 + i, +16
      v8bf b0 = *(const v8bf*)(vp + kc * 32 + hi8);
      v8bf b1 = *(const v8bf*)(vp + kc * 32 + 16 + hi8);
      acc = wmma_bf16(ae[kc], pack88(b0, b1), acc);
    }
    float* op = out + ((size_t)bb * NDIM + (c * NHEAD + hh)) * NPIX;
#pragma unroll
    for (int r = 0; r < 8; ++r)
      op[qbase + r + hi8] = acc[r] * rsinv[r];
  }
}

// ---------------------------------------------------------------------------
extern "C" void kernel_launch(void* const* d_in, const int* in_sizes, int n_in,
                              void* d_out, int out_size, void* d_ws, size_t ws_size,
                              hipStream_t stream) {
  const float* x  = (const float*)d_in[0];
  const float* Wq = (const float*)d_in[1];
  const float* bq = (const float*)d_in[2];
  const float* Wk = (const float*)d_in[3];
  const float* bk = (const float*)d_in[4];
  const float* Wv = (const float*)d_in[5];
  const float* bv = (const float*)d_in[6];
  float* out = (float*)d_out;

  char* p = (char*)d_ws;
  bf16* qbf = (bf16*)p;  p += (size_t)NBATCH * NHEAD * NPIX * NC * sizeof(bf16); // 32 MB
  bf16* kbf = (bf16*)p;  p += (size_t)NBATCH * NHEAD * NKEY * NC * sizeof(bf16); // 512 KB
  bf16* vTbf = (bf16*)p;                                                        // 512 KB
  (void)ws_size; (void)in_sizes; (void)n_in; (void)out_size;

  conv_q_kernel <<<1024,  128, 0, stream>>>(x, Wq, bq, qbf);
  conv_kv_kernel<<<512,   128, 0, stream>>>(x, Wk, bk, Wv, bv, kbf, vTbf);
  attn_kernel   <<<16384,  64, 0, stream>>>(qbf, kbf, vTbf, out);
}
